// TransitionDown_1881195676254
// MI455X (gfx1250) — matrix operand order repaired and checked by reading, compiled, not run on hardware
//
#include <hip/hip_runtime.h>
#include <hip/hip_bf16.h>
#include <stddef.h>

// ---------------- problem constants (from reference) ----------------
#define BATCH   8
#define NPTS    16384            // points per cloud
#define CIN     128              // input feature channels
#define OUT     256              // output channels
#define MPTS    (NPTS / 4)       // 4096 points kept per cloud (STRIDE=4)
#define KNN     16               // neighbors
#define FDIM    131              // 3 + CIN
#define FPAD    160              // padded to multiple of 32 for WMMA K
#define NQ      (BATCH * MPTS)   // 32768 total queries
#define NROWS   ((float)(NQ * KNN))  // BN row count = 524288
#define BN_EPS  1e-5f
#define PPT     16               // points per thread in FPS (NPTS / 1024)

// ---------------- d_out layout (floats) ----------------
#define OUT_OFF   0                      // out      [NQ*OUT]
#define NP_OFF    (NQ * OUT)             // new_pos  [NQ*3]
#define NB_OFF    (NP_OFF + NQ * 3)      // new_batch[NQ]

// ---------------- d_ws layout (bytes) ----------------
#define WS_WB     0                      // bf16 W padded [OUT][FPAD]   (81920 B)
#define WS_KNN    131072                 // int  knn idx  [NQ][KNN]     (2 MB)
#define WS_SUM    (WS_KNN + NQ * KNN * 4)     // f32 [2*OUT] sum, sumsq
#define WS_SCALE  (WS_SUM + 2 * OUT * 4)      // f32 [OUT]
#define WS_SHIFT  (WS_SCALE + OUT * 4)        // f32 [OUT]

typedef __attribute__((ext_vector_type(16))) __bf16   v16bf;
typedef __attribute__((ext_vector_type(8)))  float    v8f;

struct U8x32 { uint4 lo, hi; };          // 32-byte carrier for a WMMA bf16 fragment

__device__ __forceinline__ unsigned short f2bf(float f) {
    unsigned u = __float_as_uint(f);
    u += 0x7FFFu + ((u >> 16) & 1u);      // round-to-nearest-even
    return (unsigned short)(u >> 16);
}

// ================= kernel 0: convert W -> padded bf16, zero BN accumulators ====
__global__ __launch_bounds__(256) void k_prep(const float* __restrict__ W,
                                              unsigned short* __restrict__ Wb,
                                              float* __restrict__ sums) {
    int t = blockIdx.x * 256 + threadIdx.x;
    if (t < OUT * FPAD) {
        int o = t / FPAD, f = t - o * FPAD;
        float v = (f < FDIM) ? W[o * FDIM + f] : 0.0f;
        Wb[t] = f2bf(v);
    }
    if (t < 2 * OUT) sums[t] = 0.0f;     // must re-zero every launch
}

// ================= kernel 1: farthest point sampling (1 block per cloud) =======
// All state in registers: each of 1024 threads owns 16 points (coords + dist).
// Per iteration: unrolled local argmax + branchless coord select, 5-step
// shfl_xor wave reduction carrying (d, idx, x, y, z), 2 barriers, LDS broadcast.
__global__ __launch_bounds__(1024) void k_fps(const float* __restrict__ pos,
                                              float* __restrict__ new_pos,
                                              float* __restrict__ new_batch) {
    __shared__ float wwD[32], wwX[32], wwY[32], wwZ[32];
    __shared__ int   wwI[32];
    __shared__ float bX, bY, bZ;
    const int b = blockIdx.x, t = threadIdx.x;
    const int wv = t >> 5, lane = t & 31;
    const float* pb = pos + (size_t)b * NPTS * 3;

    float px[PPT], py[PPT], pz[PPT], dist[PPT];
    const float p0x = pb[0], p0y = pb[1], p0z = pb[2];
#pragma unroll
    for (int k = 0; k < PPT; ++k) {
        const int i = t + k * 1024;
        const float x = pb[3 * i], y = pb[3 * i + 1], z = pb[3 * i + 2];
        px[k] = x; py[k] = y; pz[k] = z;
        const float dx = x - p0x, dy = y - p0y, dz = z - p0z;
        dist[k] = dx * dx + dy * dy + dz * dz;
    }
    if (t == 0) {
        new_pos[(size_t)(b * MPTS) * 3 + 0] = p0x;
        new_pos[(size_t)(b * MPTS) * 3 + 1] = p0y;
        new_pos[(size_t)(b * MPTS) * 3 + 2] = p0z;
    }
    for (int i = t; i < MPTS; i += 1024) new_batch[b * MPTS + i] = (float)b;

    for (int it = 1; it < MPTS; ++it) {
        // local argmax over the 16 register-resident points (tie -> lowest idx)
        float d = -1.0f; int idx = 0; float cx = 0.f, cy = 0.f, cz = 0.f;
#pragma unroll
        for (int k = 0; k < PPT; ++k) {
            const bool m = dist[k] > d;
            d   = m ? dist[k] : d;
            idx = m ? (t + k * 1024) : idx;
            cx  = m ? px[k] : cx;
            cy  = m ? py[k] : cy;
            cz  = m ? pz[k] : cz;
        }
        // wave32 reduction (no memory, no barriers)
#pragma unroll
        for (int s = 16; s > 0; s >>= 1) {
            const float od = __shfl_xor(d, s, 32);
            const int   oi = __shfl_xor(idx, s, 32);
            const float ox = __shfl_xor(cx, s, 32);
            const float oy = __shfl_xor(cy, s, 32);
            const float oz = __shfl_xor(cz, s, 32);
            const bool take = (od > d) || (od == d && oi < idx);
            if (take) { d = od; idx = oi; cx = ox; cy = oy; cz = oz; }
        }
        if (lane == 0) { wwD[wv] = d; wwI[wv] = idx; wwX[wv] = cx; wwY[wv] = cy; wwZ[wv] = cz; }
        __syncthreads();
        if (wv == 0) {                    // wave 0 reduces the 32 per-wave winners
            d = wwD[lane]; idx = wwI[lane]; cx = wwX[lane]; cy = wwY[lane]; cz = wwZ[lane];
#pragma unroll
            for (int s = 16; s > 0; s >>= 1) {
                const float od = __shfl_xor(d, s, 32);
                const int   oi = __shfl_xor(idx, s, 32);
                const float ox = __shfl_xor(cx, s, 32);
                const float oy = __shfl_xor(cy, s, 32);
                const float oz = __shfl_xor(cz, s, 32);
                const bool take = (od > d) || (od == d && oi < idx);
                if (take) { d = od; idx = oi; cx = ox; cy = oy; cz = oz; }
            }
            if (lane == 0) {
                bX = cx; bY = cy; bZ = cz;
                new_pos[(size_t)(b * MPTS + it) * 3 + 0] = cx;
                new_pos[(size_t)(b * MPTS + it) * 3 + 1] = cy;
                new_pos[(size_t)(b * MPTS + it) * 3 + 2] = cz;
            }
        }
        __syncthreads();
        const float sx = bX, sy = bY, sz = bZ;
#pragma unroll
        for (int k = 0; k < PPT; ++k) {
            const float dx = px[k] - sx, dy = py[k] - sy, dz = pz[k] - sz;
            const float dd = dx * dx + dy * dy + dz * dz;
            dist[k] = dd < dist[k] ? dd : dist[k];
        }
        // no 3rd barrier needed: next bX write happens only after the next
        // __syncthreads(), and every thread has read bX before reaching it.
    }
}

// ================= kernel 2: KNN, one wave32 per query =========================
// The block's 8 waves all query the same cloud; stage its positions once in
// LDS (SoA, 192 KB) so the 8x full-cloud scans hit LDS instead of global.
__global__ __launch_bounds__(256) void k_knn(const float* __restrict__ pos,
                                             const float* __restrict__ new_pos,
                                             int* __restrict__ knn) {
    extern __shared__ float sp[];                  // X[NPTS] | Y[NPTS] | Z[NPTS]
    __shared__ float sD[8][32 * KNN];
    __shared__ int   sI[8][32 * KNN];
    const int w = threadIdx.x >> 5, lane = threadIdx.x & 31;
    const int q = blockIdx.x * 8 + w;
    const int b = q >> 12;                         // whole block is one cloud
    const float* pb = pos + (size_t)b * NPTS * 3;

    for (int i = threadIdx.x; i < NPTS; i += 256) {
        sp[i]            = pb[3 * i];
        sp[NPTS + i]     = pb[3 * i + 1];
        sp[2 * NPTS + i] = pb[3 * i + 2];
    }
    __syncthreads();

    const float qx = new_pos[(size_t)q * 3 + 0];
    const float qy = new_pos[(size_t)q * 3 + 1];
    const float qz = new_pos[(size_t)q * 3 + 2];

    float bd[KNN]; int bi[KNN];
#pragma unroll
    for (int j = 0; j < KNN; ++j) { bd[j] = 3.4e38f; bi[j] = 0; }

    for (int i = lane; i < NPTS; i += 32) {        // lane-consecutive: no bank conflicts
        const float dx = sp[i] - qx, dy = sp[NPTS + i] - qy, dz = sp[2 * NPTS + i] - qz;
        const float d = dx * dx + dy * dy + dz * dz;
        if (d < bd[KNN - 1]) {                     // branchless sorted insert
#pragma unroll
            for (int j = KNN - 1; j > 0; --j) {
                const bool sh = d < bd[j - 1];
                const bool in = d < bd[j];
                const float nd = sh ? bd[j - 1] : (in ? d : bd[j]);
                const int   ni = sh ? bi[j - 1] : (in ? i : bi[j]);
                bd[j] = nd; bi[j] = ni;
            }
            if (d < bd[0]) { bd[0] = d; bi[0] = i; }
        }
    }
#pragma unroll
    for (int j = 0; j < KNN; ++j) { sD[w][lane * KNN + j] = bd[j]; sI[w][lane * KNN + j] = bi[j]; }
    int ptr = 0;
    for (int r = 0; r < KNN; ++r) {
        float hd = (ptr < KNN) ? sD[w][lane * KNN + ptr] : 3.4e38f;
        int   hi = (ptr < KNN) ? sI[w][lane * KNN + ptr] : 0;
        int   hl = lane;
#pragma unroll
        for (int s = 16; s > 0; s >>= 1) {
            const float od = __shfl_xor(hd, s, 32);
            const int   oi = __shfl_xor(hi, s, 32);
            const int   ol = __shfl_xor(hl, s, 32);
            if (od < hd) { hd = od; hi = oi; hl = ol; }
        }
        if (lane == 0) knn[(size_t)q * KNN + r] = hi;
        if (lane == hl) ptr++;
    }
}

// ============ fused gather + WMMA GEMM body (one query per wave) ===============
__device__ __forceinline__ void gemm_query(int q, int lane,
                                           const float* __restrict__ x,
                                           const float* __restrict__ pos,
                                           const int* __restrict__ knn,
                                           const unsigned short* __restrict__ Wb,
                                           unsigned short* __restrict__ As,  // [16*FPAD] this wave
                                           v8f acc[16]) {
    const int b = q >> 12;
    int myN = 0;
    if (lane < KNN) myN = knn[(size_t)q * KNN + lane];

    // gather 16 neighbor rows -> LDS bf16 tile [16][FPAD]
    for (int r = 0; r < KNN; ++r) {
        const int nb = __shfl(myN, r, 32);
        const float* xr = x + ((size_t)b * NPTS + nb) * CIN;
        const float* pr = pos + ((size_t)b * NPTS + nb) * 3;
        for (int col = lane; col < FPAD; col += 32) {
            const float v = (col < 3) ? pr[col] : ((col < FDIM) ? xr[col - 3] : 0.0f);
            As[r * FPAD + col] = f2bf(v);
        }
    }
    // wave-private tile; same-wave LDS ordering suffices (no barrier needed)

    const int Mr = lane & 15, g = lane >> 4;
#pragma unroll
    for (int nt = 0; nt < 16; ++nt) acc[nt] = v8f{};

    // Keep rolled: full unroll makes the compiler keep 5 A-fragments + dozens
    // of B-fragments live on top of the 128-VGPR accumulator array -> spills.
#pragma unroll 1
    for (int ks = 0; ks < FPAD / 32; ++ks) {
        // A fragment: two contiguous 16B-aligned chunks from LDS (ds_load_b128 x2)
        U8x32 araw;
        araw.lo = *(const uint4*)&As[Mr * FPAD + ks * 32 + 8 * g];
        araw.hi = *(const uint4*)&As[Mr * FPAD + ks * 32 + 16 + 8 * g];
        const v16bf af = __builtin_bit_cast(v16bf, araw);
#pragma unroll
        for (int nt = 0; nt < 16; ++nt) {
            const int o = nt * 16 + Mr;          // lane's output channel within tile
            // B fragment: two 16B-aligned chunks matching A's k order (global_load_b128 x2)
            const unsigned short* bp = &Wb[(size_t)o * FPAD + ks * 32 + 8 * g];
            U8x32 braw; braw.lo = *(const uint4*)bp; braw.hi = *(const uint4*)(bp + 16);
            const v16bf bf = __builtin_bit_cast(v16bf, braw);
            acc[nt] = __builtin_amdgcn_wmma_f32_16x16x32_bf16(
                false, af, false, bf, (short)0, acc[nt], false, false);
        }
    }
}

// ================= kernel 3: pass B1 — accumulate BN statistics ================
__global__ __launch_bounds__(256) void k_stat(const float* __restrict__ x,
                                              const float* __restrict__ pos,
                                              const int* __restrict__ knn,
                                              const unsigned short* __restrict__ Wb,
                                              float* __restrict__ sums) {
    __shared__ unsigned short As[8][16 * FPAD];
    __shared__ float blkS[OUT], blkQ[OUT];
    const int w = threadIdx.x >> 5, lane = threadIdx.x & 31;
    for (int i = threadIdx.x; i < OUT; i += 256) { blkS[i] = 0.0f; blkQ[i] = 0.0f; }
    __syncthreads();

    const int nwaves = gridDim.x * 8;
    for (int q = blockIdx.x * 8 + w; q < NQ; q += nwaves) {
        v8f acc[16];
        gemm_query(q, lane, x, pos, knn, Wb, As[w], acc);
#pragma unroll
        for (int nt = 0; nt < 16; ++nt) {
            float s = 0.0f, ss = 0.0f;
#pragma unroll
            for (int v = 0; v < 8; ++v) { const float y = acc[nt][v]; s += y; ss += y * y; }
            s += __shfl_xor(s, 16, 32);
            ss += __shfl_xor(ss, 16, 32);
            if (lane < 16) {
                atomicAdd(&blkS[nt * 16 + lane], s);
                atomicAdd(&blkQ[nt * 16 + lane], ss);
            }
        }
    }
    __syncthreads();
    for (int i = threadIdx.x; i < OUT; i += 256) {
        atomicAdd(&sums[i], blkS[i]);
        atomicAdd(&sums[OUT + i], blkQ[i]);
    }
}

// ================= kernel 4: finalize BN scale/shift ===========================
__global__ __launch_bounds__(256) void k_bn(const float* __restrict__ sums,
                                            const float* __restrict__ gamma,
                                            const float* __restrict__ beta,
                                            float* __restrict__ scale,
                                            float* __restrict__ shift) {
    const int o = threadIdx.x;
    if (o < OUT) {
        const float mean = sums[o] / NROWS;
        const float var  = sums[OUT + o] / NROWS - mean * mean;
        const float sc   = gamma[o] * rsqrtf(var + BN_EPS);
        scale[o] = sc;
        shift[o] = beta[o] - mean * sc;
    }
}

// ================= kernel 5: pass B2 — recompute y, BN+ReLU+maxpool ============
__global__ __launch_bounds__(256) void k_out(const float* __restrict__ x,
                                             const float* __restrict__ pos,
                                             const int* __restrict__ knn,
                                             const unsigned short* __restrict__ Wb,
                                             const float* __restrict__ scale,
                                             const float* __restrict__ shift,
                                             float* __restrict__ out) {
    __shared__ unsigned short As[8][16 * FPAD];
    const int w = threadIdx.x >> 5, lane = threadIdx.x & 31;
    const int nwaves = gridDim.x * 8;
    for (int q = blockIdx.x * 8 + w; q < NQ; q += nwaves) {
        v8f acc[16];
        gemm_query(q, lane, x, pos, knn, Wb, As[w], acc);
        const int n = lane & 15;
#pragma unroll
        for (int nt = 0; nt < 16; ++nt) {
            const int chan = nt * 16 + n;
            const float sc = scale[chan], sh = shift[chan];
            float mx = 0.0f;                      // post-ReLU values are >= 0
#pragma unroll
            for (int v = 0; v < 8; ++v) {
                const float y = fmaxf(acc[nt][v] * sc + sh, 0.0f);
                mx = fmaxf(mx, y);
            }
            mx = fmaxf(mx, __shfl_xor(mx, 16, 32));
            if (lane < 16) out[(size_t)q * OUT + chan] = mx;
        }
    }
}

// =============================== launcher ======================================
extern "C" void kernel_launch(void* const* d_in, const int* in_sizes, int n_in,
                              void* d_out, int out_size, void* d_ws, size_t ws_size,
                              hipStream_t stream) {
    const float* x     = (const float*)d_in[0];
    const float* pos   = (const float*)d_in[1];
    // d_in[2] = batch (int32) — layout is implied, not needed
    const float* W     = (const float*)d_in[3];
    const float* gamma = (const float*)d_in[4];
    const float* beta  = (const float*)d_in[5];
    (void)in_sizes; (void)n_in; (void)out_size; (void)ws_size;

    float* outF      = (float*)d_out;
    float* new_pos   = outF + NP_OFF;
    float* new_batch = outF + NB_OFF;

    char* ws = (char*)d_ws;
    unsigned short* Wb = (unsigned short*)(ws + WS_WB);
    int*   knn   = (int*)(ws + WS_KNN);
    float* sums  = (float*)(ws + WS_SUM);
    float* scale = (float*)(ws + WS_SCALE);
    float* shift = (float*)(ws + WS_SHIFT);

    // 0) W -> bf16 padded, zero BN accumulators (every launch: deterministic)
    k_prep<<<(OUT * FPAD + 255) / 256, 256, 0, stream>>>(W, Wb, sums);
    // 1) FPS per cloud; register-resident points, 2 barriers per iteration
    k_fps<<<BATCH, 1024, 0, stream>>>(pos, new_pos, new_batch);
    // 2) KNN: one wave32 per query; cloud positions staged in 192 KB of LDS
    k_knn<<<NQ / 8, 256, 3 * NPTS * sizeof(float), stream>>>(pos, new_pos, knn);
    // 3) pass B1: gather + WMMA, accumulate per-channel sum / sumsq
    k_stat<<<1024, 256, 0, stream>>>(x, pos, knn, Wb, sums);
    // 4) BN scale/shift
    k_bn<<<1, 256, 0, stream>>>(sums, gamma, beta, scale, shift);
    // 5) pass B2: gather + WMMA + BN + ReLU + max-pool -> out
    k_out<<<1024, 256, 0, stream>>>(x, pos, knn, Wb, scale, shift, outF + OUT_OFF);
}
